// GraphAttention_31233002176554
// MI455X (gfx1250) — matrix operand order repaired and checked by reading, compiled, not run on hardware
//
#include <hip/hip_runtime.h>
#include <hip/hip_bf16.h>

// ---------------------------------------------------------------------------
// GAT layer as fused flash-attention on gfx1250 WMMA (wave32).
//   feats = X@W (per head)         -> small WMMA GEMM
//   S = (feats*a) @ feats^T        -> v_wmma_f32_16x16x32_f16, fused
//   leaky_relu, adjacency bitmask, online softmax, O += P@V  (all fused)
// Adjacency (268MB f32, binary) packed to 8MB bitmask -> L2 resident.
// K/V tiles staged to LDS with global_load_async_to_lds_b128 (ASYNCcnt),
// double-buffered, shared by 4 waves per workgroup (4x L2 traffic cut).
// ---------------------------------------------------------------------------

typedef __attribute__((ext_vector_type(16))) _Float16 v16h;
typedef __attribute__((ext_vector_type(8)))  _Float16 v8h;
typedef __attribute__((ext_vector_type(8)))  float    v8f;

#define N_NODES 8192
#define F_IN    256
#define H_HEADS 2
#define F_OUT   64
#define AW      (N_NODES / 32)      // 256 mask words per adjacency row
#define NEG_MASKV (-1.0e10f)
#define SLOPE     0.2f
#define TILE_J    64                // columns per pipelined stage
#define NT        (N_NODES / TILE_J)

// ---------------------------------------------------------------------------
// Kernel 1: pack dense binary adjacency into a bitmask (32x traffic cut).
// ---------------------------------------------------------------------------
__global__ __launch_bounds__(256) void pack_adj_kernel(
    const float* __restrict__ A, unsigned* __restrict__ bits) {
  long long idx = (long long)blockIdx.x * blockDim.x + threadIdx.x;
  float v = A[idx];
  unsigned long long m = __ballot(v > 0.5f);          // wave32 -> low 32 bits
  if ((threadIdx.x & 31) == 0) bits[idx >> 5] = (unsigned)m;
}

// ---------------------------------------------------------------------------
// Kernel 2: feats = X@W per head via WMMA; emit Q (=feats*a), K, V^T in f16.
// One wave -> one 16x64 feats tile of one head.
// A-operand (16x32 f16): lane L holds row M=L%16;
//   elem e: K = k0 + 16*(L/16)+e
// B-operand (32x16 f16): lane L holds col N=L%16; elem e: K = 16*(L/16)+e
// ---------------------------------------------------------------------------
__global__ __launch_bounds__(32) void feats_wmma_kernel(
    const float* __restrict__ X, const float* __restrict__ W,
    const float* __restrict__ avec,
    _Float16* __restrict__ fq, _Float16* __restrict__ fk,
    _Float16* __restrict__ fvt) {
  const int lane = threadIdx.x & 31;
  const int l16  = lane & 15;
  const int half = lane >> 4;
  const int head  = blockIdx.x & 1;
  const int itile = blockIdx.x >> 1;
  const int i_base = itile * 16;

  const float* Wh = W + head * (F_IN * F_OUT);

  v8f acc[4];
#pragma unroll
  for (int t = 0; t < 4; ++t)
#pragma unroll
    for (int e = 0; e < 8; ++e) acc[t][e] = 0.0f;

  const int arow = i_base + l16;
  for (int k0 = 0; k0 < F_IN; k0 += 32) {
    v16h xa;
    const float* xp = X + (long long)arow * F_IN + k0;
#pragma unroll
    for (int e = 0; e < 8; ++e) {
      xa[e]     = (_Float16)xp[16 * half + e];
      xa[8 + e] = (_Float16)xp[16 * half + 8 + e];
    }
#pragma unroll
    for (int t = 0; t < 4; ++t) {
      v16h wb;
      const int n = t * 16 + l16;
#pragma unroll
      for (int e = 0; e < 16; ++e) {
        const int k = k0 + 16 * half + e;
        wb[e] = (_Float16)Wh[k * F_OUT + n];
      }
      acc[t] = __builtin_amdgcn_wmma_f32_16x16x32_f16(
          false, xa, false, wb, (short)0, acc[t], false, false);
    }
  }

  const float* ah = avec + head * F_OUT;
#pragma unroll
  for (int t = 0; t < 4; ++t) {
    const int o = t * 16 + l16;
    const float am = ah[o];
#pragma unroll
    for (int v = 0; v < 8; ++v) {
      const int row = i_base + v + 8 * half;
      const float f = acc[t][v];
      const long long base = ((long long)head * N_NODES + row) * F_OUT + o;
      fk[base] = (_Float16)f;
      fq[base] = (_Float16)(f * am);
      fvt[((long long)head * F_OUT + o) * N_NODES + row] = (_Float16)f;
    }
  }
}

// ---------------------------------------------------------------------------
// Kernel 3: fused flash-attention GAT.
// Block = 128 threads (4 waves) = 64 rows of one head; wave w owns rows
// [w*16, w*16+16). All waves share 64x64 K/V tiles staged in LDS by async
// loads (double buffered). Per stage: 8 WMMAs for S, online softmax,
// 8 WMMAs for O.
// ---------------------------------------------------------------------------
__global__ __launch_bounds__(128) void gat_flash_kernel(
    const _Float16* __restrict__ fq, const _Float16* __restrict__ fk,
    const _Float16* __restrict__ fvt, const unsigned* __restrict__ abits,
    const float* __restrict__ bvec, float* __restrict__ out) {
  __shared__ _Float16 kbuf[2][TILE_J * F_OUT];   // [j][feat]   8KB each
  __shared__ _Float16 vbuf[2][F_OUT * TILE_J];   // [o][j]      8KB each
  __shared__ _Float16 pbuf[4][16 * TILE_J];      // per-wave P  2KB each

  const int tid  = threadIdx.x;
  const int lane = tid & 31;
  const int wv   = tid >> 5;                     // wave id 0..3
  const int l16  = lane & 15;
  const int half = lane >> 4;
  const int head   = blockIdx.x & 1;
  const int itile  = blockIdx.x >> 1;            // 0..127
  const int i_base = itile * 64 + wv * 16;       // this wave's 16 rows

  const _Float16* fqh = fq  + (long long)head * N_NODES * F_OUT;
  const _Float16* fkh = fk  + (long long)head * N_NODES * F_OUT;
  const _Float16* fvh = fvt + (long long)head * F_OUT * N_NODES;

  // ---- async stage of one 64-col tile of K and V into LDS buffer b ----
  auto issue_tile = [&](int jt, int b) {
    // K tile: 64 rows x 64 f16, contiguous 8KB copy
    {
      const _Float16* src = fkh + (long long)jt * TILE_J * F_OUT;
      unsigned dst = (unsigned)(size_t)(&kbuf[b][0]);
#pragma unroll
      for (int q = 0; q < 4; ++q) {
        const int off = (q * 128 + tid) * 8;     // f16 units, 16B chunks
        asm volatile("global_load_async_to_lds_b128 %0, %1, off"
                     :: "v"(dst + off * 2), "v"(src + off) : "memory");
      }
    }
    // V tile: 64 o-rows, each a 128B segment of a 16KB-strided row of V^T
    {
      unsigned dst = (unsigned)(size_t)(&vbuf[b][0]);
#pragma unroll
      for (int q = 0; q < 4; ++q) {
        const int e   = q * 128 + tid;           // 0..511
        const int row = e >> 3;
        const int seg = e & 7;
        const _Float16* src =
            fvh + (long long)row * N_NODES + jt * TILE_J + seg * 8;
        asm volatile("global_load_async_to_lds_b128 %0, %1, off"
                     :: "v"(dst + (row * TILE_J + seg * 8) * 2), "v"(src)
                     : "memory");
      }
    }
  };

  // ---- Q as two A-operands (feature dims 0..31 / 32..63) ----
  v16h qa0, qa1;
  {
    const _Float16* qp = fqh + (long long)(i_base + l16) * F_OUT;
#pragma unroll
    for (int e = 0; e < 8; ++e) {
      qa0[e]     = qp[16 * half + e];
      qa0[8 + e] = qp[16 * half + 8 + e];
      qa1[e]     = qp[32 + 16 * half + e];
      qa1[8 + e] = qp[40 + 16 * half + e];
    }
  }

  float m[8], l[8];
  v8f o_acc[4];
#pragma unroll
  for (int v = 0; v < 8; ++v) { m[v] = -3.0e38f; l[v] = 0.0f; }
#pragma unroll
  for (int t = 0; t < 4; ++t)
#pragma unroll
    for (int e = 0; e < 8; ++e) o_acc[t][e] = 0.0f;

  issue_tile(0, 0);

  for (int jt = 0; jt < NT; ++jt) {
    const int b = jt & 1;
    if (jt + 1 < NT) {
      issue_tile(jt + 1, b ^ 1);
      asm volatile("s_wait_asynccnt 0x8" ::: "memory");  // tile jt done
    } else {
      asm volatile("s_wait_asynccnt 0x0" ::: "memory");
    }
    __syncthreads();                    // all waves' async data visible

    // ------- S = Q @ K^T : four 16x16 C tiles (64 columns) -------
    v8f s[4];
#pragma unroll
    for (int c = 0; c < 4; ++c) {
#pragma unroll
      for (int e = 0; e < 8; ++e) s[c][e] = 0.0f;
      const _Float16* kr = &kbuf[b][(c * 16 + l16) * F_OUT + 16 * half];
      s[c] = __builtin_amdgcn_wmma_f32_16x16x32_f16(
          false, qa0, false, *(const v16h*)(kr), (short)0, s[c], false, false);
      s[c] = __builtin_amdgcn_wmma_f32_16x16x32_f16(
          false, qa1, false, *(const v16h*)(kr + 32), (short)0, s[c], false, false);
    }

    // ------- adjacency bits: two u32 words cover the 64 columns -------
    uint2 w[8];
#pragma unroll
    for (int v = 0; v < 8; ++v)
      w[v] = *(const uint2*)(abits +
              (long long)(i_base + v + 8 * half) * AW + jt * 2);

    // ------- leaky_relu + mask + online softmax over 64 columns -------
#pragma unroll
    for (int v = 0; v < 8; ++v) {
      float x[4];
#pragma unroll
      for (int c = 0; c < 4; ++c) {
        float xv = s[c][v];
        xv = xv >= 0.0f ? xv : SLOPE * xv;
        const unsigned word = (c < 2) ? w[v].x : w[v].y;
        const int bit = (word >> ((c & 1) * 16 + l16)) & 1u;
        x[c] = bit ? xv : xv + NEG_MASKV;
      }
      float t = fmaxf(fmaxf(x[0], x[1]), fmaxf(x[2], x[3]));
#pragma unroll
      for (int d = 1; d < 16; d <<= 1) t = fmaxf(t, __shfl_xor(t, d, 32));
      const float mn = fmaxf(m[v], t);
      const float sc = __expf(m[v] - mn);
      m[v] = mn;
      float e0 = __expf(x[0] - mn), e1 = __expf(x[1] - mn);
      float e2 = __expf(x[2] - mn), e3 = __expf(x[3] - mn);
      float rs = (e0 + e1) + (e2 + e3);
#pragma unroll
      for (int d = 1; d < 16; d <<= 1) rs += __shfl_xor(rs, d, 32);
      l[v] = l[v] * sc + rs;
#pragma unroll
      for (int t2 = 0; t2 < 4; ++t2) o_acc[t2][v] *= sc;

      const int prow = (v + 8 * half) * TILE_J;
      pbuf[wv][prow + l16]      = (_Float16)e0;
      pbuf[wv][prow + 16 + l16] = (_Float16)e1;
      pbuf[wv][prow + 32 + l16] = (_Float16)e2;
      pbuf[wv][prow + 48 + l16] = (_Float16)e3;
    }
    asm volatile("" ::: "memory");      // order P store -> P A-operand load

    // ------- P (C layout) -> two A operands via per-wave LDS bounce -------
    v16h pa0, pa1;
    {
      const _Float16* pr = &pbuf[wv][l16 * TILE_J];
      const v8h a0 = *(const v8h*)(pr + 16 * half);
      const v8h a1 = *(const v8h*)(pr + 8 + 16 * half);
      const v8h a2 = *(const v8h*)(pr + 32 + 16 * half);
      const v8h a3 = *(const v8h*)(pr + 40 + 16 * half);
#pragma unroll
      for (int e = 0; e < 8; ++e) {
        pa0[e] = a0[e]; pa0[8 + e] = a1[e];
        pa1[e] = a2[e]; pa1[8 + e] = a3[e];
      }
    }

    // ------- O += P @ V  (K = 64 local columns, chained x2) -------
#pragma unroll
    for (int t = 0; t < 4; ++t) {
      const _Float16* vr = &vbuf[b][(t * 16 + l16) * TILE_J + 16 * half];
      o_acc[t] = __builtin_amdgcn_wmma_f32_16x16x32_f16(
          false, pa0, false, *(const v16h*)(vr), (short)0, o_acc[t], false, false);
      o_acc[t] = __builtin_amdgcn_wmma_f32_16x16x32_f16(
          false, pa1, false, *(const v16h*)(vr + 32), (short)0, o_acc[t], false, false);
    }
    __syncthreads();                    // safe to overwrite consumed buffer
  }

  // ------- finalize: /l, +b, relu, concat heads -------
  const float* bh = bvec + head * F_OUT;
#pragma unroll
  for (int t = 0; t < 4; ++t) {
    const int o = t * 16 + l16;
    const float bb = bh[o];
#pragma unroll
    for (int v = 0; v < 8; ++v) {
      const int row = i_base + v + 8 * half;
      float val = o_acc[t][v] / l[v] + bb;
      val = val > 0.0f ? val : 0.0f;
      out[(long long)row * (H_HEADS * F_OUT) + head * F_OUT + o] = val;
    }
  }
}

// ---------------------------------------------------------------------------
extern "C" void kernel_launch(void* const* d_in, const int* in_sizes, int n_in,
                              void* d_out, int out_size, void* d_ws, size_t ws_size,
                              hipStream_t stream) {
  (void)in_sizes; (void)n_in; (void)out_size; (void)ws_size;
  const float* X = (const float*)d_in[0];
  const float* A = (const float*)d_in[1];
  const float* W = (const float*)d_in[2];
  const float* b = (const float*)d_in[3];
  const float* a = (const float*)d_in[4];

  char* ws = (char*)d_ws;
  _Float16* fq    = (_Float16*)(ws);                    // 2 MB
  _Float16* fk    = (_Float16*)(ws + (2ull << 20));     // 2 MB
  _Float16* fvt   = (_Float16*)(ws + (4ull << 20));     // 2 MB
  unsigned* abits = (unsigned*)(ws + (6ull << 20));     // 8 MB

  const long long total = (long long)N_NODES * N_NODES;
  pack_adj_kernel<<<dim3((unsigned)(total / 256)), 256, 0, stream>>>(A, abits);

  feats_wmma_kernel<<<dim3((N_NODES / 16) * H_HEADS), 32, 0, stream>>>(
      X, W, a, fq, fk, fvt);

  gat_flash_kernel<<<dim3((N_NODES / 64) * H_HEADS), 128, 0, stream>>>(
      fq, fk, fvt, abits, b, (float*)d_out);
}
